// EnhancedCell_40638980554910
// MI455X (gfx1250) — matrix operand order repaired and checked by reading, not compiled, not run on hardware
//
#include <hip/hip_runtime.h>
#include <hip/hip_bf16.h>
#include <math.h>

typedef _Float16 v8h  __attribute__((ext_vector_type(8)));
typedef _Float16 v16h __attribute__((ext_vector_type(16)));
typedef float    v8f  __attribute__((ext_vector_type(8)));

#define TAG_ 32
#define P_   2
#define NP_  2
#define NN_  2
#define E_   64
#define H_   256
#define D_   768
#define B_   64
#define S_   512
#define L_   (NP_ + S_ + NN_)      /* 516 */
#define NTOK (B_ * S_)             /* 32768 */

// ---------------------------------------------------------------------------
// WMMA fragment loaders (wave32; this code's contiguous k order, same for A and B)
// A (16 tokens x 32 K), row-major per-token rows:
//   lanes 0-15 : M=lane,    K = kb+0..15
//   lanes 16-31: M=lane-16, K = kb+16..31
// B (32 K x 16 N) built from row-major W (H x K): lane n holds column
//   h = nBase + (lane&15); lanes<16 K=kb+0..15, lanes>=16 K=kb+16..31
//   -> 16 consecutive halfs of W row h: one 32B load.
// ---------------------------------------------------------------------------
__device__ __forceinline__ v16h load_a_frag(const _Float16* arow, int kb, int lane) {
  int k0 = kb + ((lane < 16) ? 0 : 16);
  v8h a0 = *(const v8h*)(arow + k0);
  v8h a1 = *(const v8h*)(arow + k0 + 8);
  return __builtin_shufflevector(a0, a1, 0,1,2,3,4,5,6,7,8,9,10,11,12,13,14,15);
}
__device__ __forceinline__ v16h load_b_frag(const _Float16* W, int K, int nBase, int kb, int lane) {
  int h  = nBase + (lane & 15);
  int k0 = kb + ((lane < 16) ? 0 : 16);
  return *(const v16h*)(W + (size_t)h * K + k0);
}

// ---------------------------------------------------------------------------
// Prep kernels
// ---------------------------------------------------------------------------
__global__ void k_init_acc(float* acc) {
  if (threadIdx.x < 2) acc[threadIdx.x] = 0.0f;
}

__global__ void k_f32_to_f16(const float* __restrict__ src, _Float16* __restrict__ dst, int n) {
  int i = blockIdx.x * blockDim.x + threadIdx.x;
  if (i < n) dst[i] = (_Float16)src[i];
}

// feat_pad (B, 516, D) in f16: [pad_bng, pad_bng, feats(if t<len) else pad_end..., pad_end, pad_end]
__global__ void k_featpad(const float* __restrict__ feats, const int* __restrict__ lengths,
                          const float* __restrict__ pad_bng, const float* __restrict__ pad_end,
                          _Float16* __restrict__ featpad) {
  size_t idx = (size_t)blockIdx.x * blockDim.x + threadIdx.x;
  int d   = (int)(idx % D_);
  int row = (int)(idx / D_);
  int b = row / L_, pos = row % L_;
  float v;
  if (pos < NP_)                       v = pad_bng[d];
  else if ((pos - NP_) < lengths[b])   v = feats[((size_t)b * S_ + (pos - NP_)) * D_ + d];
  else                                 v = pad_end[d];
  featpad[idx] = (_Float16)v;
}

// pe_in (B, S, 128) f16: [emb(id(t)), emb(id(t+1))], id(u) = u<2 ? TAG+u : labels[u-2]
__global__ void k_pein(const int* __restrict__ labels, const float* __restrict__ emb,
                       _Float16* __restrict__ pein) {
  size_t idx = (size_t)blockIdx.x * blockDim.x + threadIdx.x;
  int col = (int)(idx & 127);
  int tok = (int)(idx >> 7);
  int b = tok >> 9, t = tok & 511;
  int p = col >> 6, e = col & 63;
  int u = t + p;
  int id = (u < P_) ? (TAG_ + u) : labels[b * S_ + (u - P_)];
  if (id < 0) id = 0;
  pein[idx] = (_Float16)emb[id * E_ + e];
}

// ---------------------------------------------------------------------------
// Kernel 2: fused 4-matrix projection, y = relu(X W^T + b), f16 out.
// grid = (8192, 4), block = 128 (4 waves); wave -> one 16x16 tile.
// blockIdx.y selects matrix: 0=pe(K=128) 1=hp(K=1536) 2=hc(K=768) 3=hn(K=1536)
// Two interleaved accumulator chains (step 64 over K) so the scheduler can
// overlap loads of one chain with the WMMA of the other (partial LOADcnt
// waits instead of s_wait_loadcnt 0).
// ---------------------------------------------------------------------------
__global__ void k_gemm_relu(const _Float16* __restrict__ featpad, const _Float16* __restrict__ pein,
                            const _Float16* __restrict__ Wpe16, const _Float16* __restrict__ Whp16,
                            const _Float16* __restrict__ Whc16, const _Float16* __restrict__ Whn16,
                            const float* __restrict__ b_pe, const float* __restrict__ b_hp,
                            const float* __restrict__ b_hc, const float* __restrict__ b_hn,
                            _Float16* __restrict__ peh, _Float16* __restrict__ hph,
                            _Float16* __restrict__ hch, _Float16* __restrict__ hnh) {
  int lane = threadIdx.x & 31;
  int tile = blockIdx.x * 4 + (threadIdx.x >> 5);
  int mat  = blockIdx.y;
  int nTile = tile & 15, mTile = tile >> 4;
  int tokBase = mTile * 16;
  int token = tokBase + (lane & 15);      // A-fragment row for this lane
  int b = token >> 9, t = token & 511;

  const _Float16* arow; const _Float16* W; const float* bias; _Float16* out; int K;
  if (mat == 0)      { arow = pein   + (size_t)token * (P_ * E_);             W = Wpe16; bias = b_pe; out = peh; K = P_ * E_;  }
  else if (mat == 1) { arow = featpad + (size_t)(b * L_ + t) * D_;            W = Whp16; bias = b_hp; out = hph; K = NP_ * D_; }
  else if (mat == 2) { arow = featpad + (size_t)(b * L_ + t + NP_) * D_;      W = Whc16; bias = b_hc; out = hch; K = D_;       }
  else               { arow = featpad + (size_t)(b * L_ + t + NP_ + 1) * D_;  W = Whn16; bias = b_hn; out = hnh; K = NN_ * D_; }

  int nBase = nTile * 16;
  v8f c0 = {}, c1 = {};
  for (int kb = 0; kb < K; kb += 64) {    // K in {128,768,1536}: all multiples of 64
    v16h a0 = load_a_frag(arow, kb, lane);
    v16h w0 = load_b_frag(W, K, nBase, kb, lane);
    c0 = __builtin_amdgcn_wmma_f32_16x16x32_f16(false, a0, false, w0, (short)0, c0, false, false);
    v16h a1 = load_a_frag(arow, kb + 32, lane);
    v16h w1 = load_b_frag(W, K, nBase, kb + 32, lane);
    c1 = __builtin_amdgcn_wmma_f32_16x16x32_f16(false, a1, false, w1, (short)0, c1, false, false);
  }
  int n = nBase + (lane & 15);
  float bv = bias[n];
  int rowtok = tokBase + ((lane < 16) ? 0 : 8);   // VGPR r -> M = r (+8 for hi lanes)
  #pragma unroll
  for (int r = 0; r < 8; ++r) {
    float v = c0[r] + c1[r] + bv;
    v = v > 0.0f ? v : 0.0f;
    out[(size_t)(rowtok + r) * H_ + n] = (_Float16)v;
  }
}

// ---------------------------------------------------------------------------
// att = softmax(W_att @ cat + b_att) over 4 heads; tiny, plain VALU.
// ---------------------------------------------------------------------------
__global__ void k_att(const _Float16* __restrict__ peh, const _Float16* __restrict__ hch,
                      const float* __restrict__ W_att, const float* __restrict__ b_att,
                      float* __restrict__ att) {
  int tok = blockIdx.x * blockDim.x + threadIdx.x;
  const _Float16* pr = peh + (size_t)tok * H_;
  const _Float16* hr = hch + (size_t)tok * H_;
  float l[4];
  #pragma unroll
  for (int j = 0; j < 4; ++j) {
    const float* w = W_att + j * (2 * H_);
    float acc = b_att[j];
    for (int k = 0; k < H_; ++k) acc += (float)pr[k] * w[k];
    for (int k = 0; k < H_; ++k) acc += (float)hr[k] * w[H_ + k];
    l[j] = acc;
  }
  float mx = fmaxf(fmaxf(l[0], l[1]), fmaxf(l[2], l[3]));
  float e0 = expf(l[0] - mx), e1 = expf(l[1] - mx), e2 = expf(l[2] - mx), e3 = expf(l[3] - mx);
  float s = e0 + e1 + e2 + e3;
  att[tok * 4 + 0] = e0 / s;
  att[tok * 4 + 1] = e1 / s;
  att[tok * 4 + 2] = e2 / s;
  att[tok * 4 + 3] = e3 / s;
}

// ---------------------------------------------------------------------------
// Kernel 3: 4 gate GEMMs on cat=[pe|hc] (K=512), one shared A fragment per
// K-step feeds 4 independent WMMA chains; epilogue fuses sigmoid + att mix.
// grid = 8192, block = 128.
// ---------------------------------------------------------------------------
__global__ void k_gates_hidden(const _Float16* __restrict__ peh, const _Float16* __restrict__ hph,
                               const _Float16* __restrict__ hch, const _Float16* __restrict__ hnh,
                               const _Float16* __restrict__ Wzpe16, const _Float16* __restrict__ Wzhp16,
                               const _Float16* __restrict__ Wzhc16, const _Float16* __restrict__ Wzhn16,
                               const float* __restrict__ bzpe, const float* __restrict__ bzhp,
                               const float* __restrict__ bzhc, const float* __restrict__ bzhn,
                               const float* __restrict__ att, _Float16* __restrict__ hiddenh) {
  int lane = threadIdx.x & 31;
  int tile = blockIdx.x * 4 + (threadIdx.x >> 5);
  int nTile = tile & 15, mTile = tile >> 4;
  int tokBase = mTile * 16;
  const _Float16* peRow = peh + (size_t)(tokBase + (lane & 15)) * H_;
  const _Float16* hcRow = hch + (size_t)(tokBase + (lane & 15)) * H_;
  int nBase = nTile * 16;

  v8f c0 = {}, c1 = {}, c2 = {}, c3 = {};
  for (int kb = 0; kb < 2 * H_; kb += 32) {
    const _Float16* arow = (kb < H_) ? peRow : (hcRow - H_);
    v16h a = load_a_frag(arow, kb, lane);
    v16h w0 = load_b_frag(Wzpe16, 2 * H_, nBase, kb, lane);
    c0 = __builtin_amdgcn_wmma_f32_16x16x32_f16(false, a, false, w0, (short)0, c0, false, false);
    v16h w1 = load_b_frag(Wzhp16, 2 * H_, nBase, kb, lane);
    c1 = __builtin_amdgcn_wmma_f32_16x16x32_f16(false, a, false, w1, (short)0, c1, false, false);
    v16h w2 = load_b_frag(Wzhc16, 2 * H_, nBase, kb, lane);
    c2 = __builtin_amdgcn_wmma_f32_16x16x32_f16(false, a, false, w2, (short)0, c2, false, false);
    v16h w3 = load_b_frag(Wzhn16, 2 * H_, nBase, kb, lane);
    c3 = __builtin_amdgcn_wmma_f32_16x16x32_f16(false, a, false, w3, (short)0, c3, false, false);
  }

  int n = nBase + (lane & 15);
  float b0 = bzpe[n], b1 = bzhp[n], b2 = bzhc[n], b3 = bzhn[n];
  int rowtok = tokBase + ((lane < 16) ? 0 : 8);
  #pragma unroll
  for (int r = 0; r < 8; ++r) {
    int tok = rowtok + r;
    float a0 = att[tok * 4 + 0], a1 = att[tok * 4 + 1], a2 = att[tok * 4 + 2], a3 = att[tok * 4 + 3];
    float s0 = 1.0f / (1.0f + expf(-(c0[r] + b0)));
    float s1 = 1.0f / (1.0f + expf(-(c1[r] + b1)));
    float s2 = 1.0f / (1.0f + expf(-(c2[r] + b2)));
    float s3 = 1.0f / (1.0f + expf(-(c3[r] + b3)));
    size_t off = (size_t)tok * H_ + n;
    float pe = (float)peh[off], hp = (float)hph[off], hc = (float)hch[off], hn = (float)hnh[off];
    float h = pe * s0 * a0 + hp * s1 * a1 + hc * s2 * a2 + hn * s3 * a3;
    hiddenh[off] = (_Float16)h;
  }
}

// ---------------------------------------------------------------------------
// Output head (32 logits, K=256) + softmax/clip/log + masked NLL reduction.
// ---------------------------------------------------------------------------
__global__ void k_out_nll(const _Float16* __restrict__ hiddenh, const float* __restrict__ W_out,
                          const float* __restrict__ b_out, const int* __restrict__ labels,
                          float* __restrict__ acc) {
  __shared__ float sh_n[256], sh_m[256];
  int tid = threadIdx.x;
  int tok = blockIdx.x * blockDim.x + tid;
  const _Float16* h = hiddenh + (size_t)tok * H_;
  float logits[TAG_];
  #pragma unroll 4
  for (int j = 0; j < TAG_; ++j) {
    const float* w = W_out + j * H_;
    float a = b_out[j];
    for (int k = 0; k < H_; ++k) a += (float)h[k] * w[k];
    logits[j] = a;
  }
  float mx = logits[0];
  #pragma unroll
  for (int j = 1; j < TAG_; ++j) mx = fmaxf(mx, logits[j]);
  float s = 0.0f;
  #pragma unroll
  for (int j = 0; j < TAG_; ++j) { logits[j] = expf(logits[j] - mx); s += logits[j]; }
  int lab = labels[tok];
  float mask = (lab != -1) ? 1.0f : 0.0f;
  int sl = lab < 0 ? 0 : lab;
  float p = fmaxf(logits[sl] / s, 1e-9f);
  float nll = -logf(p);

  sh_n[tid] = nll * mask;
  sh_m[tid] = mask;
  __syncthreads();
  for (int st = 128; st > 0; st >>= 1) {
    if (tid < st) { sh_n[tid] += sh_n[tid + st]; sh_m[tid] += sh_m[tid + st]; }
    __syncthreads();
  }
  if (tid == 0) { atomicAdd(&acc[0], sh_n[0]); atomicAdd(&acc[1], sh_m[0]); }
}

__global__ void k_final(const float* __restrict__ acc, float* __restrict__ out) {
  out[0] = acc[0] / fmaxf(acc[1], 1.0f);
}

// ---------------------------------------------------------------------------
// Launch
// ---------------------------------------------------------------------------
extern "C" void kernel_launch(void* const* d_in, const int* in_sizes, int n_in,
                              void* d_out, int out_size, void* d_ws, size_t ws_size,
                              hipStream_t stream) {
  (void)in_sizes; (void)n_in; (void)out_size; (void)ws_size;
  const float* feats    = (const float*)d_in[0];
  const int*   lengths  = (const int*)  d_in[1];
  const int*   labels   = (const int*)  d_in[2];
  const float* emb      = (const float*)d_in[3];
  const float* pad_bng  = (const float*)d_in[4];
  const float* pad_end  = (const float*)d_in[5];
  const float* W_pe  = (const float*)d_in[6];   const float* b_pe  = (const float*)d_in[7];
  const float* W_hc  = (const float*)d_in[8];   const float* b_hc  = (const float*)d_in[9];
  const float* W_hp  = (const float*)d_in[10];  const float* b_hp  = (const float*)d_in[11];
  const float* W_hn  = (const float*)d_in[12];  const float* b_hn  = (const float*)d_in[13];
  const float* Wz_pe = (const float*)d_in[14];  const float* bz_pe = (const float*)d_in[15];
  const float* Wz_hc = (const float*)d_in[16];  const float* bz_hc = (const float*)d_in[17];
  const float* Wz_hp = (const float*)d_in[18];  const float* bz_hp = (const float*)d_in[19];
  const float* Wz_hn = (const float*)d_in[20];  const float* bz_hn = (const float*)d_in[21];
  const float* W_att = (const float*)d_in[22];  const float* b_att = (const float*)d_in[23];
  const float* W_out = (const float*)d_in[24];  const float* b_out = (const float*)d_in[25];

  // ---- workspace carving (256B aligned sections) ----
  char* ws = (char*)d_ws;
  size_t cur = 0;
  auto alloc = [&](size_t bytes) -> void* {
    void* p = ws + cur;
    cur = (cur + bytes + 255) & ~(size_t)255;
    return p;
  };
  _Float16* featpad = (_Float16*)alloc((size_t)B_ * L_ * D_ * 2);      // 50.7 MB
  _Float16* pein    = (_Float16*)alloc((size_t)NTOK * P_ * E_ * 2);    //  8.4 MB
  _Float16* peh     = (_Float16*)alloc((size_t)NTOK * H_ * 2);         // 16.8 MB
  _Float16* hph     = (_Float16*)alloc((size_t)NTOK * H_ * 2);
  _Float16* hch     = (_Float16*)alloc((size_t)NTOK * H_ * 2);
  _Float16* hnh     = (_Float16*)alloc((size_t)NTOK * H_ * 2);
  _Float16* hiddenh = (_Float16*)alloc((size_t)NTOK * H_ * 2);
  _Float16* Wpe16   = (_Float16*)alloc((size_t)H_ * P_ * E_ * 2);
  _Float16* Whp16   = (_Float16*)alloc((size_t)H_ * NP_ * D_ * 2);
  _Float16* Whc16   = (_Float16*)alloc((size_t)H_ * D_ * 2);
  _Float16* Whn16   = (_Float16*)alloc((size_t)H_ * NN_ * D_ * 2);
  _Float16* Wzpe16  = (_Float16*)alloc((size_t)H_ * 2 * H_ * 2);
  _Float16* Wzhp16  = (_Float16*)alloc((size_t)H_ * 2 * H_ * 2);
  _Float16* Wzhc16  = (_Float16*)alloc((size_t)H_ * 2 * H_ * 2);
  _Float16* Wzhn16  = (_Float16*)alloc((size_t)H_ * 2 * H_ * 2);
  float*    att     = (float*)   alloc((size_t)NTOK * 4 * 4);
  float*    acc     = (float*)   alloc(2 * sizeof(float));

  // ---- prep ----
  k_init_acc<<<1, 32, 0, stream>>>(acc);
  {
    size_t total = (size_t)B_ * L_ * D_;                 // 25,362,432 = 256 * 99072
    k_featpad<<<(int)(total / 256), 256, 0, stream>>>(feats, lengths, pad_bng, pad_end, featpad);
  }
  k_pein<<<NTOK * P_ * E_ / 256, 256, 0, stream>>>(labels, emb, pein);

  auto conv = [&](const float* src, _Float16* dst, int n) {
    k_f32_to_f16<<<(n + 255) / 256, 256, 0, stream>>>(src, dst, n);
  };
  conv(W_pe,  Wpe16,  H_ * P_ * E_);
  conv(W_hp,  Whp16,  H_ * NP_ * D_);
  conv(W_hc,  Whc16,  H_ * D_);
  conv(W_hn,  Whn16,  H_ * NN_ * D_);
  conv(Wz_pe, Wzpe16, H_ * 2 * H_);
  conv(Wz_hp, Wzhp16, H_ * 2 * H_);
  conv(Wz_hc, Wzhc16, H_ * 2 * H_);
  conv(Wz_hn, Wzhn16, H_ * 2 * H_);

  // ---- main WMMA GEMMs ----
  {
    // 2048 M-tiles * 16 N-tiles = 32768 wave-tiles per matrix; 4 waves/block
    dim3 grid(NTOK / 16 * (H_ / 16) / 4, 4);
    k_gemm_relu<<<grid, 128, 0, stream>>>(featpad, pein,
                                          Wpe16, Whp16, Whc16, Whn16,
                                          b_pe, b_hp, b_hc, b_hn,
                                          peh, hph, hch, hnh);
  }
  k_att<<<NTOK / 256, 256, 0, stream>>>(peh, hch, W_att, b_att, att);
  k_gates_hidden<<<NTOK / 16 * (H_ / 16) / 4, 128, 0, stream>>>(peh, hph, hch, hnh,
                                                                Wzpe16, Wzhp16, Wzhc16, Wzhn16,
                                                                bz_pe, bz_hp, bz_hc, bz_hn,
                                                                att, hiddenh);
  k_out_nll<<<NTOK / 256, 256, 0, stream>>>(hiddenh, W_out, b_out, labels, acc);
  k_final<<<1, 1, 0, stream>>>(acc, (float*)d_out);
}
